// GraphAttention_5007931867443
// MI455X (gfx1250) — matrix operand order repaired and checked by reading, not compiled, not run on hardware
//
#include <hip/hip_runtime.h>
#include <hip/hip_bf16.h>

typedef __bf16 bf16;
typedef __attribute__((ext_vector_type(16))) __bf16 v16bf;
typedef __attribute__((ext_vector_type(8)))  __bf16 v8bf;
typedef __attribute__((ext_vector_type(8)))  float  v8f;

#define NPOINT  1024
#define NSAMPLE 32
#define DCH     128
#define KDIM    131          // D + 3
#define KPAD    160          // 5 * 32 (bf16 wmma k-step = 32)
#define KSTEPS  5
#define NTILES  8
#define ALPHA   0.2f
#define EPS     1e-4f

#define BFRAG_ELEMS (KSTEPS * NTILES * 32 * 16)   // 20480 bf16 = 40 KB

// ---------------------------------------------------------------------------
// Pre-swizzle the weight matrix a[131][128] (fp32) into WMMA B-fragment order
// (bf16, zero-padded to K=160). Layout: frag[((kstep*8+ntile)*32+lane)*16+e]
// where B-fragment element (lane,e) holds B[k][n], k = (e&7) + 8*(lane/16) +
// 16*(e/8) + 32*kstep, n = ntile*16 + lane%16  (CDNA5 16-bit B 32x16 layout).
// ---------------------------------------------------------------------------
__global__ void prep_bfrag(const float* __restrict__ a, bf16* __restrict__ bfrag) {
    int idx = blockIdx.x * blockDim.x + threadIdx.x;
    if (idx >= BFRAG_ELEMS) return;
    int e     = idx & 15;
    int lane  = (idx >> 4) & 31;
    int ntile = (idx >> 9) & 7;
    int kstep = idx >> 12;
    int k = (e & 7) + 8 * (lane >> 4) + 16 * (e >> 3) + 32 * kstep;
    int n = ntile * 16 + (lane & 15);
    float v = (k < KDIM) ? a[k * DCH + n] : 0.0f;
    bfrag[idx] = (bf16)v;
}

// ---------------------------------------------------------------------------
// Main kernel: one 256-thread block (8 wave32) per (b, n) point.
// ---------------------------------------------------------------------------
__global__ __launch_bounds__(256) void graph_attn_main(
    const float* __restrict__ grouped,   // [B,np,ns,D]
    const float* __restrict__ delta,     // [B,np,ns,D]
    const float* __restrict__ dxyz,      // [B,np,ns,3]
    const bf16*  __restrict__ bfrag,     // pre-swizzled a
    const float* __restrict__ bparam,    // [2,1,np,D]
    const float* __restrict__ cparam,    // [1,np,D]
    const float* __restrict__ convf_w, const float* __restrict__ convf_b,
    const float* __restrict__ convd_w, const float* __restrict__ convd_b,
    const float* __restrict__ bnf_g, const float* __restrict__ bnf_b,
    const float* __restrict__ bnd_g, const float* __restrict__ bnd_b,
    float* __restrict__ out)             // [B,np,D]
{
    __shared__ bf16  A_lds[NSAMPLE * KPAD];        // 10240 B : x = [delta|xyz|0]
    __shared__ float e_lds[NSAMPLE * DCH];         // 16 KB  : e, then exp(e-m)
    __shared__ float delta_lds[NSAMPLE * DCH];     // 16 KB  : delta fp32 reuse
    __shared__ float xyz_lds[NSAMPLE * 4];         // stride-4 padded
    __shared__ float partf[NSAMPLE * 8];
    __shared__ float df_lds[NSAMPLE], fd_lds[NSAMPLE];
    __shared__ float pmax[2 * DCH], psum[2 * DCH];
    __shared__ float pgp1[2 * DCH], pgp2[2 * DCH];

    const int t   = threadIdx.x;
    const int blk = blockIdx.x;                    // b*NPOINT + n
    const int n   = blk & (NPOINT - 1);
    const size_t base = (size_t)blk * (NSAMPLE * DCH);

    // ---- load phase: delta -> LDS (fp32 + bf16), xyz, zero pad ----
    for (int i = t; i < NSAMPLE * DCH; i += 256) {
        float v = delta[base + i];
        delta_lds[i] = v;
        int s = i >> 7, d = i & 127;
        A_lds[s * KPAD + d] = (bf16)v;
    }
    for (int i = t; i < NSAMPLE * (KPAD - KDIM); i += 256) {   // cols 131..159
        int s = i / (KPAD - KDIM), c = i % (KPAD - KDIM);
        A_lds[s * KPAD + KDIM + c] = (bf16)0.0f;
    }
    if (t < NSAMPLE * 3) {                                     // cols 128..130
        int s = t / 3, c = t - 3 * s;
        float v = dxyz[(size_t)blk * (NSAMPLE * 3) + t];
        xyz_lds[s * 4 + c] = v;
        A_lds[s * KPAD + DCH + c] = (bf16)v;
    }
    __syncthreads();

    // ---- WMMA: e[32x128] = x[32x160] @ a[160x128], bf16 in / f32 acc ----
    const int wave = t >> 5;           // = N-tile
    const int lane = t & 31;
    const int row   = lane & 15;
    const int khalf = (lane >> 4) << 3;   // 0 or 8 (A-layout half)
    v8f acc0 = {}; v8f acc1 = {};
    const v16bf* bvec = (const v16bf*)bfrag;
    for (int ks = 0; ks < KSTEPS; ++ks) {
        v16bf bf = bvec[(ks * NTILES + wave) * 32 + lane];
        int kb = khalf + 32 * ks;
        union { v16bf v; v8bf h[2]; } ua0, ua1;
        ua0.h[0] = *(const v8bf*)&A_lds[row * KPAD + kb];
        ua0.h[1] = *(const v8bf*)&A_lds[row * KPAD + kb + 16];
        ua1.h[0] = *(const v8bf*)&A_lds[(row + 16) * KPAD + kb];
        ua1.h[1] = *(const v8bf*)&A_lds[(row + 16) * KPAD + kb + 16];
        acc0 = __builtin_amdgcn_wmma_f32_16x16x32_bf16(
            false, ua0.v, false, bf, (short)0, acc0, false, false);
        acc1 = __builtin_amdgcn_wmma_f32_16x16x32_bf16(
            false, ua1.v, false, bf, (short)0, acc1, false, false);
    }
    // store with LeakyReLU; C/D layout: elem v -> row v+8*(lane>=16), col lane%16
    {
        int col    = wave * 16 + (lane & 15);
        int rowoff = (lane >> 4) * 8;
        for (int v = 0; v < 8; ++v) {
            float x0 = acc0[v]; x0 = x0 > 0.f ? x0 : ALPHA * x0;
            float x1 = acc1[v]; x1 = x1 > 0.f ? x1 : ALPHA * x1;
            e_lds[(rowoff + v) * DCH + col]        = x0;
            e_lds[(16 + rowoff + v) * DCH + col]   = x1;
        }
    }

    // ---- f gate partials: dot(delta[s,:], convf_w) split 8 ways ----
    {
        int s = t >> 3, p = t & 7;
        float sum = 0.f;
        const float* dl = &delta_lds[s * DCH + p * 16];
        const float* wf = &convf_w[p * 16];
        #pragma unroll
        for (int j = 0; j < 16; ++j) sum += dl[j] * wf[j];
        partf[s * 8 + p] = sum;
    }
    __syncthreads();

    // ---- wave 0: f/d softmaxes over 32 samples, df/fd normalization ----
    if (t < 32) {
        int s = t;
        float fsum = 0.f;
        #pragma unroll
        for (int p = 0; p < 8; ++p) fsum += partf[s * 8 + p];
        const float bn_scale = rsqrtf(1.0f + 1e-5f);
        float flin = (fsum + convf_b[0]) * bn_scale * bnf_g[0] + bnf_b[0];
        float dlin = (xyz_lds[s * 4 + 0] * convd_w[0] +
                      xyz_lds[s * 4 + 1] * convd_w[1] +
                      xyz_lds[s * 4 + 2] * convd_w[2] + convd_b[0])
                     * bn_scale * bnd_g[0] + bnd_b[0];
        float m = flin;
        for (int o = 16; o; o >>= 1) m = fmaxf(m, __shfl_xor(m, o, 32));
        float ef = __expf(flin - m), se = ef;
        for (int o = 16; o; o >>= 1) se += __shfl_xor(se, o, 32);
        float fval = ef / se + 1.0f;
        float m2 = dlin;
        for (int o = 16; o; o >>= 1) m2 = fmaxf(m2, __shfl_xor(m2, o, 32));
        float ed = __expf(dlin - m2), sd = ed;
        for (int o = 16; o; o >>= 1) sd += __shfl_xor(sd, o, 32);
        float dval = ed / sd + 1.0f;
        float df = (dval + EPS) / (fval + EPS);
        float fd = (fval + EPS) / (dval + EPS);
        float mdf = df, mfd = fd;
        for (int o = 16; o; o >>= 1) { mdf += __shfl_xor(mdf, o, 32);
                                       mfd += __shfl_xor(mfd, o, 32); }
        mdf *= (1.0f / 32.0f); mfd *= (1.0f / 32.0f);
        df_lds[s] = df / (mdf + EPS);
        fd_lds[s] = fd / (mfd + EPS);
    }
    __syncthreads();

    // ---- e softmax over s + weighted sums; 2 threads per feature column ----
    const int d  = t & 127;
    const int h  = t >> 7;
    const int s0 = h * 16;
    {   // pass 1: partial max
        float m = -1e30f;
        for (int s = s0; s < s0 + 16; ++s) m = fmaxf(m, e_lds[s * DCH + d]);
        pmax[h * DCH + d] = m;
    }
    __syncthreads();
    {   // pass 2: exp in place + partial sum
        float m = fmaxf(pmax[d], pmax[DCH + d]);
        float sum = 0.f;
        for (int s = s0; s < s0 + 16; ++s) {
            float ex = __expf(e_lds[s * DCH + d] - m);
            e_lds[s * DCH + d] = ex;
            sum += ex;
        }
        psum[h * DCH + d] = sum;
    }
    __syncthreads();
    {   // pass 3: gp1 = sum df*att1*grouped ; gp2 = sum fd*(1-att1)*delta
        float inv = 1.0f / (psum[d] + psum[DCH + d]);
        float gp1 = 0.f, gp2 = 0.f;
        for (int s = s0; s < s0 + 16; ++s) {
            float p  = e_lds[s * DCH + d] * inv;
            float a1 = df_lds[s] * p;
            float a2 = fd_lds[s] * (1.0f - p);
            gp1 += a1 * grouped[base + (size_t)s * DCH + d];   // coalesced HBM
            gp2 += a2 * delta_lds[s * DCH + d];
        }
        pgp1[h * DCH + d] = gp1;
        pgp2[h * DCH + d] = gp2;
    }
    __syncthreads();
    if (t < DCH) {
        float gp1 = pgp1[t] + pgp1[DCH + t];
        float gp2 = pgp2[t] + pgp2[DCH + t];
        float b0 = bparam[n * DCH + t];
        float b1 = bparam[NPOINT * DCH + n * DCH + t];
        float w0 = 1.0f / (1.0f + __expf(b1 - b0));    // softmax over 2
        out[(size_t)blk * DCH + t] = w0 * gp1 + (1.0f - w0) * gp2
                                     + cparam[n * DCH + t];
    }
}

// ---------------------------------------------------------------------------
extern "C" void kernel_launch(void* const* d_in, const int* in_sizes, int n_in,
                              void* d_out, int out_size, void* d_ws, size_t ws_size,
                              hipStream_t stream) {
    // input order: center_feature(0, unused), grouped_feature(1), delta(2),
    // delta_xyz(3), a(4), b(5), c(6), convf_w(7), convf_b(8), convd_w(9),
    // convd_b(10), bnf_gamma(11), bnf_beta(12), bnd_gamma(13), bnd_beta(14)
    const float* grouped = (const float*)d_in[1];
    const float* delta   = (const float*)d_in[2];
    const float* dxyz    = (const float*)d_in[3];
    const float* a       = (const float*)d_in[4];
    const float* bparam  = (const float*)d_in[5];
    const float* cparam  = (const float*)d_in[6];
    const float* convf_w = (const float*)d_in[7];
    const float* convf_b = (const float*)d_in[8];
    const float* convd_w = (const float*)d_in[9];
    const float* convd_b = (const float*)d_in[10];
    const float* bnf_g   = (const float*)d_in[11];
    const float* bnf_b   = (const float*)d_in[12];
    const float* bnd_g   = (const float*)d_in[13];
    const float* bnd_b   = (const float*)d_in[14];

    bf16* bfrag = (bf16*)d_ws;   // 40 KB of workspace

    prep_bfrag<<<(BFRAG_ELEMS + 255) / 256, 256, 0, stream>>>(a, bfrag);
    graph_attn_main<<<8 * NPOINT, 256, 0, stream>>>(
        grouped, delta, dxyz, bfrag, bparam, cparam,
        convf_w, convf_b, convd_w, convd_b,
        bnf_g, bnf_b, bnd_g, bnd_b, (float*)d_out);
}
